// DynamicLinear_56599079027182
// MI455X (gfx1250) — matrix operand order repaired and checked by reading, not compiled, not run on hardware
//
#include <hip/hip_runtime.h>

// Problem constants (match reference)
#define L_DIM   1024
#define B_DIM   8
#define C_IN    1024
#define C_OUT   1024
#define C_COND  256
#define K_EXP   4
#define HID     64
#define TEMP    30.0f

typedef __attribute__((ext_vector_type(16))) _Float16 v16h;
typedef __attribute__((ext_vector_type(8)))  _Float16 v8h;
typedef __attribute__((ext_vector_type(4)))  _Float16 v4h;
typedef __attribute__((ext_vector_type(8)))  float    v8f;
typedef __attribute__((ext_vector_type(4)))  float    v4f;

union HFrag {
    v16h v;
    v8h  h[2];
};

// ---------------------------------------------------------------------------
// Kernel 1: attention path (tiny). One block, 256 threads.
// Computes att (B x K) and b_b (B x C_OUT) into workspace.
// ---------------------------------------------------------------------------
__global__ void attention_kernel(const float* __restrict__ cond,
                                 const float* __restrict__ fc1_w,
                                 const float* __restrict__ fc1_b,
                                 const float* __restrict__ fc2_w,
                                 const float* __restrict__ fc2_b,
                                 const float* __restrict__ bias,
                                 float* __restrict__ att_g,
                                 float* __restrict__ bb_g) {
    __shared__ float s_h[B_DIM * HID];
    __shared__ float s_att[B_DIM * K_EXP];
    const int t = threadIdx.x;

    // h = relu(cond @ fc1_w.T + fc1_b): 8*64 = 512 outputs
    for (int idx = t; idx < B_DIM * HID; idx += 256) {
        const int b = idx >> 6;       // /HID
        const int j = idx & (HID - 1);
        float s = fc1_b[j];
        const float* cr = cond + b * C_COND;
        const float* wr = fc1_w + j * C_COND;
        #pragma unroll 4
        for (int c = 0; c < C_COND; ++c) s += cr[c] * wr[c];
        s_h[idx] = fmaxf(s, 0.0f);
    }
    __syncthreads();

    // logits = (h @ fc2_w.T + fc2_b) / TEMP : 8*4 = 32 outputs
    if (t < B_DIM * K_EXP) {
        const int b = t >> 2;
        const int k = t & 3;
        float s = fc2_b[k];
        const float* hr = s_h + b * HID;
        const float* wr = fc2_w + k * HID;
        #pragma unroll
        for (int j = 0; j < HID; ++j) s += hr[j] * wr[j];
        s_att[t] = s * (1.0f / TEMP);
    }
    __syncthreads();

    // softmax over K per batch
    if (t < B_DIM) {
        float m = s_att[t * K_EXP];
        #pragma unroll
        for (int k = 1; k < K_EXP; ++k) m = fmaxf(m, s_att[t * K_EXP + k]);
        float e[K_EXP];
        float sum = 0.0f;
        #pragma unroll
        for (int k = 0; k < K_EXP; ++k) { e[k] = __expf(s_att[t * K_EXP + k] - m); sum += e[k]; }
        const float inv = 1.0f / sum;
        #pragma unroll
        for (int k = 0; k < K_EXP; ++k) {
            const float v = e[k] * inv;
            s_att[t * K_EXP + k] = v;
            att_g[t * K_EXP + k] = v;
        }
    }
    __syncthreads();

    // b_b = att @ bias : 8 * 1024 outputs
    for (int idx = t; idx < B_DIM * C_OUT; idx += 256) {
        const int b = idx >> 10;
        const int o = idx & (C_OUT - 1);
        float s = 0.0f;
        #pragma unroll
        for (int k = 0; k < K_EXP; ++k) s += s_att[b * K_EXP + k] * bias[k * C_OUT + o];
        bb_g[idx] = s;
    }
}

// ---------------------------------------------------------------------------
// Kernel 2: convert x (L,B,C_IN) f32 -> xh[b][l][i] f16 (per-batch row-major)
// ---------------------------------------------------------------------------
__global__ void convert_x_kernel(const float* __restrict__ x,
                                 _Float16* __restrict__ xh) {
    const size_t flat = ((size_t)blockIdx.x * blockDim.x + threadIdx.x) * 4;
    if (flat >= (size_t)L_DIM * B_DIM * C_IN) return;
    const int    i    = (int)(flat & (C_IN - 1));
    const size_t rest = flat >> 10;          // / C_IN
    const int    b    = (int)(rest & (B_DIM - 1));
    const size_t l    = rest >> 3;           // / B_DIM
    const v4f xv = *(const v4f*)(x + flat);
    v4h o;
    #pragma unroll
    for (int q = 0; q < 4; ++q) o[q] = (_Float16)xv[q];
    *(v4h*)(xh + ((size_t)b * L_DIM + l) * C_IN + i) = o;
}

// ---------------------------------------------------------------------------
// Kernel 3: w_b[b][o][i] = f16( sum_k att[b][k] * weight[k][o][i] )
// Each thread handles 4 contiguous i; reads weight once, writes all 8 batches.
// ---------------------------------------------------------------------------
__global__ void combine_w_kernel(const float* __restrict__ weight,
                                 const float* __restrict__ att_g,
                                 _Float16* __restrict__ wh) {
    __shared__ float s_att[B_DIM * K_EXP];
    if (threadIdx.x < B_DIM * K_EXP) s_att[threadIdx.x] = att_g[threadIdx.x];
    __syncthreads();

    const size_t idx = ((size_t)blockIdx.x * blockDim.x + threadIdx.x) * 4;
    if (idx >= (size_t)C_OUT * C_IN) return;

    v4f wv[K_EXP];
    #pragma unroll
    for (int k = 0; k < K_EXP; ++k)
        wv[k] = *(const v4f*)(weight + (size_t)k * C_OUT * C_IN + idx);

    #pragma unroll
    for (int b = 0; b < B_DIM; ++b) {
        v4f acc = s_att[b * K_EXP + 0] * wv[0];
        #pragma unroll
        for (int k = 1; k < K_EXP; ++k) acc += s_att[b * K_EXP + k] * wv[k];
        v4h o;
        #pragma unroll
        for (int q = 0; q < 4; ++q) o[q] = (_Float16)acc[q];
        *(v4h*)(wh + (size_t)b * C_OUT * C_IN + idx) = o;
    }
}

// ---------------------------------------------------------------------------
// Kernel 4: batched GEMM via WMMA f32_16x16x32_f16.
//   out[l][b][o] = sum_i xh[b][l][i] * wh[b][o][i] + bb[b][o]
// Grid: (N/128, M/128, B). Block: 256 threads = 8 waves, waves in 2x4:
//   wave (wm, wn) owns a 64(M) x 32(N) tile -> 4x2 WMMA accumulators.
// Fragments loaded straight from global per the CDNA5 wave32 layouts;
// the 32MB f16 working set lives in the 192MB L2, so panel re-reads hit L2.
// ---------------------------------------------------------------------------
__global__ void __launch_bounds__(256)
gemm_wmma_kernel(const _Float16* __restrict__ xh,
                 const _Float16* __restrict__ wh,
                 const float* __restrict__ bb_g,
                 float* __restrict__ out) {
    const int b    = blockIdx.z;
    const int M0   = blockIdx.y * 128;
    const int N0   = blockIdx.x * 128;
    const int lane = threadIdx.x & 31;
    const int wave = threadIdx.x >> 5;
    const int wm   = wave >> 2;  // 0..1 -> 64-row group
    const int wn   = wave & 3;   // 0..3 -> 32-col group

    const _Float16* A = xh + (size_t)b * L_DIM * C_IN;   // (L x C_IN) row-major
    const _Float16* Bm = wh + (size_t)b * C_OUT * C_IN;  // (C_OUT x C_IN) row-major

    const int grp   = lane >> 4;     // 0: lanes 0-15, 1: lanes 16-31
    const int lane16 = lane & 15;

    // Per-lane base pointers (row for A fragments, column-row for B fragments)
    const _Float16* arow[4];
    #pragma unroll
    for (int mt = 0; mt < 4; ++mt) {
        const int row = M0 + wm * 64 + mt * 16 + lane16;
        arow[mt] = A + (size_t)row * C_IN;
    }
    const _Float16* brow[2];
    #pragma unroll
    for (int nt = 0; nt < 2; ++nt) {
        const int col = N0 + wn * 32 + nt * 16 + lane16;
        brow[nt] = Bm + (size_t)col * C_IN;
    }

    v8f acc[4][2];
    #pragma unroll
    for (int mt = 0; mt < 4; ++mt)
        #pragma unroll
        for (int nt = 0; nt < 2; ++nt)
            acc[mt][nt] = (v8f)(0.0f);

    for (int k0 = 0; k0 < C_IN; k0 += 32) {
        // A fragments: 16x32 f16. lanes<16: K = {k0..k0+7, k0+16..k0+23};
        // lanes>=16:  K = {k0+8..k0+15, k0+24..k0+31}.
        HFrag a[4];
        #pragma unroll
        for (int mt = 0; mt < 4; ++mt) {
            const _Float16* pa = arow[mt] + k0 + grp * 8;
            a[mt].h[0] = *(const v8h*)(pa);
            a[mt].h[1] = *(const v8h*)(pa + 16);
            __builtin_prefetch(pa + 32, 0, 1);   // next K slice -> global_prefetch_b8
        }
        // B fragments: 32x16 f16, column = lane16; lanes<16: K = {k0..k0+7, k0+16..k0+23};
        // lanes>=16: K = {k0+8..k0+15, k0+24..k0+31} (two b128 loads).
        HFrag bf[2];
        #pragma unroll
        for (int nt = 0; nt < 2; ++nt) {
            const _Float16* pb = brow[nt] + k0 + grp * 8;
            bf[nt].h[0] = *(const v8h*)(pb);
            bf[nt].h[1] = *(const v8h*)(pb + 16);
            __builtin_prefetch(pb + 32, 0, 1);
        }
        // 8 WMMAs per wave per K-step
        #pragma unroll
        for (int mt = 0; mt < 4; ++mt)
            #pragma unroll
            for (int nt = 0; nt < 2; ++nt)
                acc[mt][nt] = __builtin_amdgcn_wmma_f32_16x16x32_f16(
                    /*neg_a=*/false, a[mt].v,
                    /*neg_b=*/false, bf[nt].v,
                    /*c_mod=*/(short)0, acc[mt][nt],
                    /*reuse_a=*/false, /*reuse_b=*/false);
    }

    // Epilogue: C/D layout — VGPR r: M=r (lanes 0-15) / M=r+8 (lanes 16-31),
    // N = lane16. Fuse +b_b.
    #pragma unroll
    for (int nt = 0; nt < 2; ++nt) {
        const int o  = N0 + wn * 32 + nt * 16 + lane16;
        const float bbv = bb_g[b * C_OUT + o];
        #pragma unroll
        for (int mt = 0; mt < 4; ++mt) {
            const int mbase = M0 + wm * 64 + mt * 16 + grp * 8;
            #pragma unroll
            for (int r = 0; r < 8; ++r) {
                const int m = mbase + r;
                out[((size_t)m * B_DIM + b) * C_OUT + o] = acc[mt][nt][r] + bbv;
            }
        }
    }
}

// ---------------------------------------------------------------------------
// Launch
// ---------------------------------------------------------------------------
extern "C" void kernel_launch(void* const* d_in, const int* in_sizes, int n_in,
                              void* d_out, int out_size, void* d_ws, size_t ws_size,
                              hipStream_t stream) {
    const float* x      = (const float*)d_in[0];
    const float* cond   = (const float*)d_in[1];
    const float* fc1_w  = (const float*)d_in[2];
    const float* fc1_b  = (const float*)d_in[3];
    const float* fc2_w  = (const float*)d_in[4];
    const float* fc2_b  = (const float*)d_in[5];
    const float* weight = (const float*)d_in[6];
    const float* bias   = (const float*)d_in[7];
    float* out = (float*)d_out;

    char* ws = (char*)d_ws;
    // Workspace layout
    float*    att_g = (float*)(ws);                                   // 128 B
    float*    bb_g  = (float*)(ws + 4096);                            // 32 KB
    _Float16* xh    = (_Float16*)(ws + (1 << 20));                    // 16 MB
    _Float16* wh    = (_Float16*)(ws + (1 << 20) + (16 << 20));       // 16 MB

    attention_kernel<<<1, 256, 0, stream>>>(cond, fc1_w, fc1_b, fc2_w, fc2_b,
                                            bias, att_g, bb_g);

    {
        const size_t n = (size_t)L_DIM * B_DIM * C_IN / 4;  // 2,097,152 threads
        convert_x_kernel<<<(unsigned)(n / 256), 256, 0, stream>>>(x, xh);
    }
    {
        const size_t n = (size_t)C_OUT * C_IN / 4;          // 262,144 threads
        combine_w_kernel<<<(unsigned)(n / 256), 256, 0, stream>>>(weight, att_g, wh);
    }

    dim3 grid(C_OUT / 128, L_DIM / 128, B_DIM);             // (8, 8, 8)
    gemm_wmma_kernel<<<grid, 256, 0, stream>>>(xh, wh, bb_g, out);
}
